// ChessMoveSelector_74552042324746
// MI455X (gfx1250) — matrix operand order repaired and checked by reading, not compiled, not run on hardware
//
#include <hip/hip_runtime.h>
#include <hip/hip_bf16.h>
#include <math.h>

typedef _Float16 half_t;
typedef __attribute__((ext_vector_type(16))) _Float16 v16h;
typedef __attribute__((ext_vector_type(8)))  float    v8f;

#define NB    4096   // batch
#define NMAX  64
#define BD    256
#define MD    128
#define NEX   3

union Frag { uint4 q[2]; v16h h; };

// ---------------------------------------------------------------------------
// Kernel 0: convert fc_w -> f16, conv2_w -> f16 rearranged [tap][oc][ic]
// ---------------------------------------------------------------------------
__global__ __launch_bounds__(256) void convert_weights(
    const float* __restrict__ fc_w,     // (256,4096)
    const float* __restrict__ conv2_w,  // (64,32,3,3)
    half_t* __restrict__ fc_wh,         // (256,4096) f16
    half_t* __restrict__ w2h)           // (9,64,32)  f16
{
  const int i = blockIdx.x * 256 + threadIdx.x;
  if (i < BD * 4096) fc_wh[i] = (half_t)fc_w[i];
  if (i < 9 * 64 * 32) {
    const int ic  = i & 31;
    const int oc  = (i >> 5) & 63;
    const int tap = i >> 11;
    // conv2_w index: ((oc*32+ic)*3+ky)*3+kx = (oc*32+ic)*9 + tap
    w2h[i] = (half_t)conv2_w[(oc * 32 + ic) * 9 + tap];
  }
}

// ---------------------------------------------------------------------------
// Kernel 1: fused conv1 (VALU f32) + conv2 (WMMA implicit GEMM), one image
// per workgroup (8 waves). Output x2: f16 (B,4096), flat = oc*64 + px.
// ---------------------------------------------------------------------------
__global__ __launch_bounds__(256) void conv_fused(
    const float* __restrict__ board,    // (B,6,8,8)
    const float* __restrict__ conv1_w,  // (32,6,3,3) = [oc1*54 + c*9 + tap]
    const float* __restrict__ conv1_b,  // (32)
    const float* __restrict__ conv2_b,  // (64)
    const half_t* __restrict__ w2h,     // (9,64,32)
    half_t* __restrict__ x2)            // (B,4096)
{
  __shared__ float  s_board[6 * 64];
  __shared__ float  s_w1[32 * 54];      // [oc1][c*9+tap]
  __shared__ half_t s_x1[65 * 32];      // [px][ic], row 64 = zeros (boundary)
  __shared__ half_t s_w2[9 * 64 * 32];  // [tap][oc][ic]
  __shared__ half_t s_x2[64 * 64];      // [oc][px]

  const int t = threadIdx.x;
  const int b = blockIdx.x;

  for (int i = t; i < 6 * 64; i += 256)  s_board[i] = board[b * 384 + i];
  for (int i = t; i < 32 * 54; i += 256) s_w1[i] = conv1_w[i];
  for (int i = t; i < (9 * 64 * 32) / 8; i += 256)
    ((uint4*)s_w2)[i] = ((const uint4*)w2h)[i];
  if (t < 32) s_x1[64 * 32 + t] = (half_t)0.0f;   // zero row for padding taps
  __syncthreads();

  // ---- conv1: each thread -> 1 pixel x 8 output channels -----------------
  {
    const int px  = t & 63;
    const int icb = (t >> 6) * 8;          // wave-uniform
    const int y = px >> 3, x = px & 7;
    float acc[8];
    for (int i = 0; i < 8; ++i) acc[i] = conv1_b[icb + i];
    for (int tap = 0; tap < 9; ++tap) {
      const int sy = y + tap / 3 - 1;
      const int sx = x + tap % 3 - 1;
      if ((unsigned)sy < 8u && (unsigned)sx < 8u) {
        const int ps = sy * 8 + sx;
        for (int c = 0; c < 6; ++c) {
          const float xv = s_board[c * 64 + ps];
          for (int i = 0; i < 8; ++i)
            acc[i] = fmaf(xv, s_w1[(icb + i) * 54 + c * 9 + tap], acc[i]);
        }
      }
    }
    for (int i = 0; i < 8; ++i)
      s_x1[px * 32 + icb + i] = (half_t)fmaxf(acc[i], 0.0f);
  }
  __syncthreads();

  // ---- conv2 as implicit GEMM: O(64px x 64oc) = sum_tap Xs(64x32)*W(32x64)
  {
    const int wv     = t >> 5;
    const int lane   = t & 31;
    const int l16    = lane & 15;
    const int half16 = lane >> 4;
    for (int i = 0; i < 2; ++i) {
      const int tid = wv * 2 + i;          // 0..15 tiles
      const int mt = tid >> 2, nt = tid & 3;
      const int px = mt * 16 + l16;        // A row (output pixel)
      const int y = px >> 3, x = px & 7;
      const int oc = nt * 16 + l16;        // B col (output channel)
      v8f acc = {};
      for (int tap = 0; tap < 9; ++tap) {
        const int sy = y + tap / 3 - 1;
        const int sx = x + tap % 3 - 1;
        const bool inb = ((unsigned)sy < 8u) && ((unsigned)sx < 8u);
        const int ps = inb ? (sy * 8 + sx) : 64;   // row 64 is all-zero
        Frag A, Bf;
        // A: halves [8h, 8h+8) and [16+8h, 16+8h+8) of row ps -> 2x b128
        A.q[0] = *(const uint4*)&s_x1[ps * 32 + 8 * half16];
        A.q[1] = *(const uint4*)&s_x1[ps * 32 + 16 + 8 * half16];
        // B: halves [8h, 8h+8) and [16+8h, 16+8h+8) of row (tap,oc) -> 2x b128
        const int wb = (tap * 64 + oc) * 32 + 8 * half16;
        Bf.q[0] = *(const uint4*)&s_w2[wb];
        Bf.q[1] = *(const uint4*)&s_w2[wb + 16];
        acc = __builtin_amdgcn_wmma_f32_16x16x32_f16(
            false, A.h, false, Bf.h, (short)0, acc, false, false);
      }
      const float bias = conv2_b[oc];
      for (int r = 0; r < 8; ++r) {
        const int m = r + 8 * half16;      // pixel row within tile
        const float v = fmaxf(acc[r] + bias, 0.0f);
        s_x2[(nt * 16 + l16) * 64 + (mt * 16 + m)] = (half_t)v;
      }
    }
  }
  __syncthreads();

  // ---- coalesced store of flat activation (oc*64+px) ---------------------
  {
    const uint4* src = (const uint4*)s_x2;         // 512 x 16B
    uint4* dst = (uint4*)(x2 + (size_t)b * 4096);
    for (int i = t; i < 512; i += 256) dst[i] = src[i];
  }
}

// ---------------------------------------------------------------------------
// Kernel 2: fc GEMM (4096x4096)x(4096x256) with WMMA; epilogue fuses
// relu(+fc_b) and the K=3 extra branch. One 16x16 tile per wave.
// ---------------------------------------------------------------------------
__global__ __launch_bounds__(256) void fc_gemm(
    const half_t* __restrict__ x2,      // (4096,4096)
    const half_t* __restrict__ fc_wh,   // (256,4096)
    const float* __restrict__ fc_b,     // (256)
    const float* __restrict__ extra,    // (4096,3)
    const float* __restrict__ extra_w,  // (256,3)
    const float* __restrict__ extra_b,  // (256)
    float* __restrict__ emb)            // (4096,256)
{
  const int t = threadIdx.x;
  const int g = blockIdx.x * 8 + (t >> 5);   // wave id: 0..4095
  const int mt = g >> 4;                     // 256 row tiles
  const int nt = g & 15;                     // 16 col tiles
  const int lane   = t & 31;
  const int l16    = lane & 15;
  const int half16 = lane >> 4;
  const int brow = mt * 16 + l16;            // A row = batch index
  const int j    = nt * 16 + l16;            // output feature
  const half_t* __restrict__ ar = x2 + (size_t)brow * 4096 + 8 * half16;
  const half_t* __restrict__ br = fc_wh + (size_t)j * 4096 + 8 * half16;

  v8f acc = {};
  for (int kk = 0; kk < 128; ++kk) {
    const int k0 = kk * 32;
    Frag A, Bf;
    A.q[0]  = *(const uint4*)&ar[k0];
    A.q[1]  = *(const uint4*)&ar[k0 + 16];
    Bf.q[0] = *(const uint4*)&br[k0];
    Bf.q[1] = *(const uint4*)&br[k0 + 16];
    acc = __builtin_amdgcn_wmma_f32_16x16x32_f16(
        false, A.h, false, Bf.h, (short)0, acc, false, false);
  }

  const float fcb = fc_b[j], eb = extra_b[j];
  const float ew0 = extra_w[j * 3 + 0];
  const float ew1 = extra_w[j * 3 + 1];
  const float ew2 = extra_w[j * 3 + 2];
  for (int r = 0; r < 8; ++r) {
    const int bi = mt * 16 + r + 8 * half16;
    const float e0 = extra[bi * 3 + 0];
    const float e1 = extra[bi * 3 + 1];
    const float e2 = extra[bi * 3 + 2];
    const float v = fmaxf(acc[r] + fcb, 0.0f) + eb + e0 * ew0 + e1 * ew1 + e2 * ew2;
    emb[(size_t)bi * 256 + j] = v;
  }
}

// ---------------------------------------------------------------------------
// Kernel 3: score + masked softmax. One wave32 per board; the move branch
// collapses to moves[b,n,:] . (move_w^T wm) + move_b . wm.
// ---------------------------------------------------------------------------
__global__ __launch_bounds__(128) void score_softmax(
    const float* __restrict__ emb,      // (4096,256)
    const float* __restrict__ moves,    // (4096,64,2)
    const int*   __restrict__ lengths,  // (4096)
    const float* __restrict__ move_w,   // (128,2)
    const float* __restrict__ move_b,   // (128)
    const float* __restrict__ comb_w,   // (1,384)
    const float* __restrict__ comb_b,   // (1)
    float* __restrict__ out)            // (4096,64)
{
  const int b = blockIdx.x * 4 + (threadIdx.x >> 5);
  const int lane = threadIdx.x & 31;

  float v20 = 0.f, v21 = 0.f, cmv = comb_b[0];
  for (int m = 0; m < MD; ++m) {
    const float wm = comb_w[BD + m];
    v20 = fmaf(move_w[m * 2 + 0], wm, v20);
    v21 = fmaf(move_w[m * 2 + 1], wm, v21);
    cmv = fmaf(move_b[m], wm, cmv);
  }
  float sb = 0.f;
  const float* __restrict__ e = emb + (size_t)b * 256;
  for (int j = 0; j < BD; ++j) sb = fmaf(e[j], comb_w[j], sb);

  const int len = lengths[b];
  float sc[2]; int valid[2];
  for (int i = 0; i < 2; ++i) {
    const int n = lane + 32 * i;
    const float m0 = moves[((size_t)b * 64 + n) * 2 + 0];
    const float m1 = moves[((size_t)b * 64 + n) * 2 + 1];
    sc[i] = sb + cmv + m0 * v20 + m1 * v21;
    valid[i] = (n < len);
  }
  float mx = fmaxf(valid[0] ? sc[0] : -1e30f, valid[1] ? sc[1] : -1e30f);
  for (int off = 16; off > 0; off >>= 1)
    mx = fmaxf(mx, __shfl_xor(mx, off, 32));
  const float e0 = valid[0] ? __expf(sc[0] - mx) : 0.f;
  const float e1 = valid[1] ? __expf(sc[1] - mx) : 0.f;
  float s = e0 + e1;
  for (int off = 16; off > 0; off >>= 1)
    s += __shfl_xor(s, off, 32);
  const float inv = 1.0f / s;
  out[(size_t)b * 64 + lane]      = e0 * inv;
  out[(size_t)b * 64 + lane + 32] = e1 * inv;
}

// ---------------------------------------------------------------------------
extern "C" void kernel_launch(void* const* d_in, const int* in_sizes, int n_in,
                              void* d_out, int out_size, void* d_ws, size_t ws_size,
                              hipStream_t stream) {
  const float* board   = (const float*)d_in[0];
  const float* extra   = (const float*)d_in[1];
  const float* moves   = (const float*)d_in[2];
  const int*   lengths = (const int*)  d_in[3];
  const float* conv1_w = (const float*)d_in[4];
  const float* conv1_b = (const float*)d_in[5];
  const float* conv2_w = (const float*)d_in[6];
  const float* conv2_b = (const float*)d_in[7];
  const float* fc_w    = (const float*)d_in[8];
  const float* fc_b    = (const float*)d_in[9];
  const float* extra_w = (const float*)d_in[10];
  const float* extra_b = (const float*)d_in[11];
  const float* move_w  = (const float*)d_in[12];
  const float* move_b  = (const float*)d_in[13];
  const float* comb_w  = (const float*)d_in[14];
  const float* comb_b  = (const float*)d_in[15];
  float* out = (float*)d_out;

  char* ws = (char*)d_ws;
  size_t off = 0;
  half_t* x2    = (half_t*)(ws + off); off += (size_t)NB * 4096 * sizeof(half_t);   // 32 MB
  half_t* fc_wh = (half_t*)(ws + off); off += (size_t)BD * 4096 * sizeof(half_t);   // 2 MB
  half_t* w2h   = (half_t*)(ws + off); off += (size_t)9 * 64 * 32 * sizeof(half_t); // 36 KB
  float*  emb   = (float*) (ws + off); off += (size_t)NB * BD * sizeof(float);      // 4 MB

  convert_weights<<<(BD * 4096 + 255) / 256, 256, 0, stream>>>(fc_w, conv2_w, fc_wh, w2h);
  conv_fused<<<NB, 256, 0, stream>>>(board, conv1_w, conv1_b, conv2_b, w2h, x2);
  fc_gemm<<<512, 256, 0, stream>>>(x2, fc_wh, fc_b, extra, extra_w, extra_b, emb);
  score_softmax<<<NB / 4, 128, 0, stream>>>(emb, moves, lengths, move_w, move_b, comb_w, comb_b, out);
}
